// PcdObjEncoder_6167573037753
// MI455X (gfx1250) — matrix operand order repaired and checked by reading, compiled, not run on hardware
//
#include <hip/hip_runtime.h>

typedef _Float16 half_t;
typedef __attribute__((ext_vector_type(16))) _Float16 v16h;
typedef __attribute__((ext_vector_type(8)))  _Float16 v8h;
typedef __attribute__((ext_vector_type(8)))  float    v8f;

#define NTHREADS 256
#define NWAVES   8

// ---------------- dynamic LDS layout (bytes) ----------------
#define OFF_XYZ    0        // 1024*3*4 = 12288
#define OFF_FEAT   12288    // 1024*3*4 = 12288
#define OFF_DIST   24576    // 1024*4   = 4096
#define OFF_FIDX1  28672    // 32*4
#define OFF_NXYZ1  28800    // 32*3*4
#define OFF_GIDX1  29184    // 32*32*4  = 4096
#define OFF_FIDX2  33280    // 16*4
#define OFF_NXYZ2  33344    // 16*3*4
#define OFF_GIDX2  33536    // 16*32*4  = 2048
#define OFF_REDV   35584    // 8*4
#define OFF_REDI   35616    // 8*4
#define OFF_OUT1   35648    // 32*128*4 = 16384
#define OFF_OUT2   52032    // 16*256*4 = 16384
#define OFF_WACT   68416    // 8 waves * (2560+2048+2048) halfs * 2B = 106496
#define OFF_A3IN   174912   // 16*288*2 = 9216
#define OFF_A3H1   184128   // 16*256*2 = 8192
#define OFF_A3H2   192320   // 16*512*2 = 16384
#define OFF_VEC    208704   // 768*4    = 3072
#define SMEM_BYTES 211776

// per-wave activation region (halfs): act0 16x160, act1 16x128, act2 16x128
#define WACT_PER_WAVE 6656

// ---------------- swizzled f16 weight offsets in d_ws (halfs) ----------------
// layout: per (kslab, ntile): 512-half block; each lane's 16 B-frag elems contiguous.
#define WSO_1A 0        // 32x64
#define WSO_1B 2048     // 64x64
#define WSO_1C 6144     // 64x128
#define WSO_2A 14336    // 160x128
#define WSO_2B 34816    // 128x128
#define WSO_2C 51200    // 128x256
#define WSO_3A 83968    // 288x256
#define WSO_3B 157696   // 256x512
#define WSO_3C 288768   // 512x768
#define WS_TOTAL 681984 // halfs (~1.36 MB)

// ---------------- WMMA helpers ----------------
__device__ __forceinline__ v8f wmma_f16(v16h a, v16h b, v8f c) {
  // D = A(16x32 f16) x B(32x16 f16) + C(16x16 f32)
  return __builtin_amdgcn_wmma_f32_16x16x32_f16(false, a, false, b, (short)0, c, false, false);
}

// A fragment from a zero-padded LDS f16 tile act[16][stride], K window [k0,k0+32).
// ISA A layout: lane(0-15) M=lane, elems 0..7 -> K k0+0..7(+8*hi), elems 8..15 -> +16.
// Two contiguous 16B runs -> 2x ds_load_b128.
__device__ __forceinline__ v16h frag_a_lds(const half_t* act, int stride, int k0, int lane) {
  int hi = (lane >> 4) & 1, m = lane & 15;
  const half_t* p = act + m * stride + k0 + hi * 8;
  v8h lo = *(const v8h*)(p);
  v8h hb = *(const v8h*)(p + 16);
  v16h a;
#pragma unroll
  for (int e = 0; e < 8; e++) { a[e] = lo[e]; a[e + 8] = hb[e]; }
  return a;
}

// B fragment from pre-swizzled f16 weights: one unguarded 32B contiguous load per lane.
__device__ __forceinline__ v16h frag_b_sw(const half_t* __restrict__ Wsw, int ntiles,
                                          int ks, int nt, int lane) {
  const half_t* p = Wsw + (((size_t)(ks * ntiles + nt)) << 9) + (lane << 4);
  return *(const v16h*)p;
}

// C/D f32 layout: VGPR r -> (M = r + 8*laneHi, N = lane&15)
__device__ __forceinline__ void epilogue_store(v8f c, const float* __restrict__ S,
                                               const float* __restrict__ Bb, int n0,
                                               half_t* act, int stride, int lane) {
  int hi = (lane >> 4) & 1, col = n0 + (lane & 15);
  float sc = S[col], bb = Bb[col];
#pragma unroll
  for (int r = 0; r < 8; r++) {
    int m = r + hi * 8;
    float v = c[r] * sc + bb;
    v = v > 0.0f ? v : 0.0f;
    act[m * stride + col] = (half_t)v;
  }
}

// relu epilogue + max over all 16 M rows -> per-lane scalar for col n0+(lane&15)
__device__ __forceinline__ float epilogue_max(v8f c, const float* __restrict__ S,
                                              const float* __restrict__ Bb, int n0, int lane) {
  int col = n0 + (lane & 15);
  float sc = S[col], bb = Bb[col];
  float mx = 0.0f;
#pragma unroll
  for (int r = 0; r < 8; r++) {
    float v = c[r] * sc + bb;
    v = v > 0.0f ? v : 0.0f;
    mx = fmaxf(mx, v);
  }
  mx = fmaxf(mx, __shfl_xor(mx, 16, 32));  // combine M 0-7 with M 8-15
  return mx;
}

// ---------------- FPS (block-wide): fidx[0]=0, argmax ties -> lowest idx ----------------
__device__ void fps_block(const float* pts, int N, int np, int* fidx, float* dist,
                          float* redv, int* redi) {
  int tid = threadIdx.x, lane = tid & 31, w = tid >> 5;
  for (int i = tid; i < N; i += NTHREADS) dist[i] = 1e10f;
  if (tid == 0) fidx[0] = 0;
  __syncthreads();
  for (int it = 0; it < np; ++it) {
    int far = fidx[it];
    float cx = pts[far * 3 + 0], cy = pts[far * 3 + 1], cz = pts[far * 3 + 2];
    float bv = -1.0f;
    int bi = 0x7fffffff;
    for (int i = tid; i < N; i += NTHREADS) {
      float dx = pts[i * 3 + 0] - cx, dy = pts[i * 3 + 1] - cy, dz = pts[i * 3 + 2] - cz;
      float d = dx * dx + dy * dy + dz * dz;
      float nd = fminf(dist[i], d);
      dist[i] = nd;
      if (nd > bv) { bv = nd; bi = i; }
    }
#pragma unroll
    for (int off = 16; off > 0; off >>= 1) {
      float ov = __shfl_xor(bv, off, 32);
      int   oi = __shfl_xor(bi, off, 32);
      if (ov > bv || (ov == bv && oi < bi)) { bv = ov; bi = oi; }
    }
    if (lane == 0) { redv[w] = bv; redi[w] = bi; }
    __syncthreads();
    if (tid == 0 && it + 1 < np) {
      float fv = redv[0]; int fi = redi[0];
      for (int j = 1; j < NWAVES; j++)
        if (redv[j] > fv || (redv[j] == fv && redi[j] < fi)) { fv = redv[j]; fi = redi[j]; }
      fidx[it + 1] = fi;
    }
    __syncthreads();
  }
}

// ---- ball query: in-radius indices in ascending order, pad with first (or 0) ----
__device__ void ball_query_block(const float* pts, int N, const float* cent, int ncent,
                                 float r2, int* gidx) {
  int tid = threadIdx.x, lane = tid & 31, w = tid >> 5;
  for (int s = w; s < ncent; s += NWAVES) {
    float cx = cent[s * 3 + 0], cy = cent[s * 3 + 1], cz = cent[s * 3 + 2];
    int cnt = 0;
    for (int base = 0; base < N; base += 32) {
      int p = base + lane;
      float dx = pts[p * 3 + 0] - cx, dy = pts[p * 3 + 1] - cy, dz = pts[p * 3 + 2] - cz;
      bool pred = (dx * dx + dy * dy + dz * dz) < r2;
      unsigned m = (unsigned)__ballot(pred);
      int slot = cnt + __popc(m & ((1u << lane) - 1u));
      if (pred && slot < 32) gidx[s * 32 + slot] = p;
      cnt += __popc(m);
      if (cnt >= 32) break;
    }
    asm volatile("s_wait_dscnt 0x0" ::: "memory");  // LDS in-order per wave; belt & braces
    int first = (cnt > 0) ? gidx[s * 32] : 0;
    if (lane >= cnt) gidx[s * 32 + lane] = first;
  }
}

// ======== weight pre-swizzle: f32 row-major (cin x cout) -> f16 fragment-blocked, K padded ======
__global__ void __launch_bounds__(NTHREADS) swizzle_weights(const float* __restrict__ src,
                                                            half_t* __restrict__ dst,
                                                            int cin, int cout, int cinPad) {
  int ntiles = cout >> 4;
  int total = cinPad * cout;
  for (int o = blockIdx.x * blockDim.x + threadIdx.x; o < total; o += gridDim.x * blockDim.x) {
    int blk = o >> 9, within = o & 511;
    int lane = within >> 4, e = within & 15;
    int ks = blk / ntiles, nt = blk - ks * ntiles;
    int k = ks * 32 + (e & 7) + ((lane >> 4) & 1) * 8 + (e >> 3) * 16;  // B-frag: elem e -> K = k0 + (e&7) + 8*laneHi + 16*(e>>3)
    int n = nt * 16 + (lane & 15);
    dst[o] = (half_t)((k < cin) ? src[k * cout + n] : 0.0f);
  }
}

// ============================ main fused kernel: one block per object ============================
__global__ void __launch_bounds__(NTHREADS) pcd_encoder_kernel(
    const float* __restrict__ pc,
    const half_t* __restrict__ W1a, const half_t* __restrict__ W1b, const half_t* __restrict__ W1c,
    const float* __restrict__ S1a, const float* __restrict__ S1b, const float* __restrict__ S1c,
    const float* __restrict__ B1a, const float* __restrict__ B1b, const float* __restrict__ B1c,
    const half_t* __restrict__ W2a, const half_t* __restrict__ W2b, const half_t* __restrict__ W2c,
    const float* __restrict__ S2a, const float* __restrict__ S2b, const float* __restrict__ S2c,
    const float* __restrict__ B2a, const float* __restrict__ B2b, const float* __restrict__ B2c,
    const half_t* __restrict__ W3a, const half_t* __restrict__ W3b, const half_t* __restrict__ W3c,
    const float* __restrict__ S3a, const float* __restrict__ S3b, const float* __restrict__ S3c,
    const float* __restrict__ B3a, const float* __restrict__ B3b, const float* __restrict__ B3c,
    const float* __restrict__ FCw, const float* __restrict__ FCb,
    float* __restrict__ out) {
  extern __shared__ char smem[];
  float*  s_xyz   = (float*)(smem + OFF_XYZ);
  float*  s_feat  = (float*)(smem + OFF_FEAT);
  float*  s_dist  = (float*)(smem + OFF_DIST);
  int*    s_fidx1 = (int*)  (smem + OFF_FIDX1);
  float*  s_nxyz1 = (float*)(smem + OFF_NXYZ1);
  int*    s_gidx1 = (int*)  (smem + OFF_GIDX1);
  int*    s_fidx2 = (int*)  (smem + OFF_FIDX2);
  float*  s_nxyz2 = (float*)(smem + OFF_NXYZ2);
  int*    s_gidx2 = (int*)  (smem + OFF_GIDX2);
  float*  s_redv  = (float*)(smem + OFF_REDV);
  int*    s_redi  = (int*)  (smem + OFF_REDI);
  float*  s_out1  = (float*)(smem + OFF_OUT1);
  float*  s_out2  = (float*)(smem + OFF_OUT2);
  half_t* s_wact  = (half_t*)(smem + OFF_WACT);
  half_t* s_a3in  = (half_t*)(smem + OFF_A3IN);
  half_t* s_a3h1  = (half_t*)(smem + OFF_A3H1);
  half_t* s_a3h2  = (half_t*)(smem + OFF_A3H2);
  float*  s_vec   = (float*)(smem + OFF_VEC);

  const int tid = threadIdx.x, lane = tid & 31, w = tid >> 5;
  const int obj = blockIdx.x;

  half_t* act0 = s_wact + (size_t)w * WACT_PER_WAVE;   // 16x160 (zero-padded grouped input)
  half_t* act1 = act0 + 16 * 160;                      // 16x128
  half_t* act2 = act1 + 16 * 128;                      // 16x128

  // ---- stage 0: load point cloud ----
  const float* basep = pc + (size_t)obj * 1024 * 6;
  for (int t = tid; t < 1024 * 6; t += NTHREADS) {
    int p = t / 6, c = t % 6;
    float v = basep[t];
    if (c < 3) s_xyz[p * 3 + c] = v; else s_feat[p * 3 + (c - 3)] = v;
  }
  __syncthreads();

  // ---- SA1: FPS + ball query ----
  fps_block(s_xyz, 1024, 32, s_fidx1, s_dist, s_redv, s_redi);
  if (tid < 96) { int m = tid / 3, c = tid % 3; s_nxyz1[tid] = s_xyz[s_fidx1[m] * 3 + c]; }
  __syncthreads();
  ball_query_block(s_xyz, 1024, s_nxyz1, 32, 0.04f, s_gidx1);
  for (int i = tid; i < 32 * 128; i += NTHREADS) s_out1[i] = 0.0f;
  __syncthreads();

  // ---- SA1 MLP [6->64->64->128] + max over 32 samples; wave w owns centers 4w..4w+3 ----
#pragma unroll 1
  for (int ci = 0; ci < 4; ci++) {
    int s = w * 4 + ci;
#pragma unroll 1
    for (int hh = 0; hh < 2; hh++) {      // two 16-row tiles per center
      int tile = s * 2 + hh;
      // materialize zero-padded grouped 16x32 tile into act0 (stride 32): lane = column
      {
        int c = lane;  // 0..31, cols >=6 are zero pad
#pragma unroll 1
        for (int i = 0; i < 16; i++) {
          int row = tile * 16 + i, s2 = row >> 5, kk = row & 31;
          int idx = s_gidx1[s2 * 32 + kk];
          float v = 0.0f;
          if (c < 3)      v = s_xyz[idx * 3 + c] - s_nxyz1[s2 * 3 + c];
          else if (c < 6) v = s_feat[idx * 3 + (c - 3)];
          act0[i * 32 + c] = (half_t)v;
        }
      }
      v16h a0 = frag_a_lds(act0, 32, 0, lane);
#pragma unroll 1
      for (int n = 0; n < 4; n++) {        // layer1: 6 -> 64
        v8f c = {};
        c = wmma_f16(a0, frag_b_sw(W1a, 4, 0, n, lane), c);
        epilogue_store(c, S1a, B1a, n * 16, act1, 64, lane);
      }
#pragma unroll 1
      for (int n = 0; n < 4; n++) {        // layer2: 64 -> 64
        v8f c = {};
#pragma unroll 1
        for (int ks = 0; ks < 2; ks++)
          c = wmma_f16(frag_a_lds(act1, 64, ks * 32, lane), frag_b_sw(W1b, 4, ks, n, lane), c);
        epilogue_store(c, S1b, B1b, n * 16, act2, 64, lane);
      }
#pragma unroll 1
      for (int n = 0; n < 8; n++) {        // layer3: 64 -> 128, fused max
        v8f c = {};
#pragma unroll 1
        for (int ks = 0; ks < 2; ks++)
          c = wmma_f16(frag_a_lds(act2, 64, ks * 32, lane), frag_b_sw(W1c, 8, ks, n, lane), c);
        float mxv = epilogue_max(c, S1c, B1c, n * 16, lane);
        if (lane < 16) {
          float* p = &s_out1[s * 128 + n * 16 + lane];
          *p = fmaxf(*p, mxv);
        }
      }
    }
  }
  __syncthreads();

  // ---- SA2: FPS + ball query over the 32 SA1 centers ----
  fps_block(s_nxyz1, 32, 16, s_fidx2, s_dist, s_redv, s_redi);
  if (tid < 48) { int m = tid / 3, c = tid % 3; s_nxyz2[tid] = s_nxyz1[s_fidx2[m] * 3 + c]; }
  __syncthreads();
  ball_query_block(s_nxyz1, 32, s_nxyz2, 16, 0.16f, s_gidx2);
  for (int i = tid; i < 16 * 256; i += NTHREADS) s_out2[i] = 0.0f;
  __syncthreads();

  // ---- SA2 MLP [131->128->128->256] + max; wave w owns centers 2w..2w+1 ----
#pragma unroll 1
  for (int ci = 0; ci < 2; ci++) {
    int s = w * 2 + ci;
#pragma unroll 1
    for (int hh = 0; hh < 2; hh++) {
      int tile = s * 2 + hh;
      // materialize zero-padded grouped 16x160 tile into act0
#pragma unroll 1
      for (int t = lane; t < 16 * 160; t += 32) {
        int m = t / 160, c = t - m * 160;
        int row = tile * 16 + m, s2 = row >> 5, kk = row & 31;
        int idx = s_gidx2[s2 * 32 + kk];
        float v = 0.0f;
        if (c < 3)        v = s_nxyz1[idx * 3 + c] - s_nxyz2[s2 * 3 + c];
        else if (c < 131) v = s_out1[idx * 128 + (c - 3)];
        act0[t] = (half_t)v;
      }
#pragma unroll 1
      for (int n = 0; n < 8; n++) {        // layer1: 131 -> 128
        v8f c = {};
#pragma unroll 1
        for (int ks = 0; ks < 5; ks++)
          c = wmma_f16(frag_a_lds(act0, 160, ks * 32, lane), frag_b_sw(W2a, 8, ks, n, lane), c);
        epilogue_store(c, S2a, B2a, n * 16, act1, 128, lane);
      }
#pragma unroll 1
      for (int n = 0; n < 8; n++) {        // layer2: 128 -> 128
        v8f c = {};
#pragma unroll 1
        for (int ks = 0; ks < 4; ks++)
          c = wmma_f16(frag_a_lds(act1, 128, ks * 32, lane), frag_b_sw(W2b, 8, ks, n, lane), c);
        epilogue_store(c, S2b, B2b, n * 16, act2, 128, lane);
      }
#pragma unroll 1
      for (int n = 0; n < 16; n++) {       // layer3: 128 -> 256, fused max
        v8f c = {};
#pragma unroll 1
        for (int ks = 0; ks < 4; ks++)
          c = wmma_f16(frag_a_lds(act2, 128, ks * 32, lane), frag_b_sw(W2c, 16, ks, n, lane), c);
        float mxv = epilogue_max(c, S2c, B2c, n * 16, lane);
        if (lane < 16) {
          float* p = &s_out2[s * 256 + n * 16 + lane];
          *p = fmaxf(*p, mxv);
        }
      }
    }
  }
  __syncthreads();

  // ---- SA3: global MLP [259->256->512->768] over 16 rows (one M-tile, waves split N) ----
  for (int t = tid; t < 16 * 288; t += NTHREADS) {
    int m = t / 288, c = t % 288;
    float v = 0.0f;
    if (c < 3)        v = s_nxyz2[m * 3 + c];
    else if (c < 259) v = s_out2[m * 256 + (c - 3)];
    s_a3in[t] = (half_t)v;
  }
  __syncthreads();
#pragma unroll 1
  for (int ni = 0; ni < 2; ni++) {         // layer1: 259 -> 256
    int n = w * 2 + ni;
    v8f c = {};
#pragma unroll 1
    for (int ks = 0; ks < 9; ks++)
      c = wmma_f16(frag_a_lds(s_a3in, 288, ks * 32, lane), frag_b_sw(W3a, 16, ks, n, lane), c);
    epilogue_store(c, S3a, B3a, n * 16, s_a3h1, 256, lane);
  }
  __syncthreads();
#pragma unroll 1
  for (int ni = 0; ni < 4; ni++) {         // layer2: 256 -> 512
    int n = w * 4 + ni;
    v8f c = {};
#pragma unroll 1
    for (int ks = 0; ks < 8; ks++)
      c = wmma_f16(frag_a_lds(s_a3h1, 256, ks * 32, lane), frag_b_sw(W3b, 32, ks, n, lane), c);
    epilogue_store(c, S3b, B3b, n * 16, s_a3h2, 512, lane);
  }
  __syncthreads();
  {                                        // layer3: 512 -> 768, K-outer, 6 accumulators
    v8f z = {};
    v8f acc[6];
#pragma unroll
    for (int i = 0; i < 6; i++) acc[i] = z;
#pragma unroll 1
    for (int ks = 0; ks < 16; ks++) {
      v16h a = frag_a_lds(s_a3h2, 512, ks * 32, lane);
#pragma unroll
      for (int ni = 0; ni < 6; ni++) {
        int n = w * 6 + ni;
        acc[ni] = wmma_f16(a, frag_b_sw(W3c, 48, ks, n, lane), acc[ni]);
      }
    }
#pragma unroll
    for (int ni = 0; ni < 6; ni++) {
      int n = w * 6 + ni;
      float mxv = epilogue_max(acc[ni], S3c, B3c, n * 16, lane);
      if (lane < 16) s_vec[n * 16 + lane] = mxv;
    }
  }
  __syncthreads();

  // ---- FC 768x768 matvec + bias (VALU; negligible FLOPs) ----
  for (int o = tid; o < 768; o += NTHREADS) {
    float accv = FCb[o];
#pragma unroll 4
    for (int c = 0; c < 768; c++) accv += s_vec[c] * FCw[c * 768 + o];
    out[(size_t)obj * 768 + o] = accv;
  }
}

extern "C" void kernel_launch(void* const* d_in, const int* in_sizes, int n_in,
                              void* d_out, int out_size, void* d_ws, size_t ws_size,
                              hipStream_t stream) {
  (void)n_in; (void)out_size; (void)ws_size;
  const float* pc = (const float*)d_in[0];
  int nobj = in_sizes[0] / (1024 * 6);  // B*O = 1280
  auto F = [&](int i) { return (const float*)d_in[i]; };
  half_t* ws = (half_t*)d_ws;

  auto sw = [&](const float* src, half_t* dst, int cin, int cout, int cinPad) {
    int total = cinPad * cout;
    int blocks = (total + NTHREADS - 1) / NTHREADS;
    swizzle_weights<<<blocks, NTHREADS, 0, stream>>>(src, dst, cin, cout, cinPad);
  };
  sw(F(1),  ws + WSO_1A, 6,   64,  32);
  sw(F(2),  ws + WSO_1B, 64,  64,  64);
  sw(F(3),  ws + WSO_1C, 64,  128, 64);
  sw(F(10), ws + WSO_2A, 131, 128, 160);
  sw(F(11), ws + WSO_2B, 128, 128, 128);
  sw(F(12), ws + WSO_2C, 128, 256, 128);
  sw(F(19), ws + WSO_3A, 259, 256, 288);
  sw(F(20), ws + WSO_3B, 256, 512, 256);
  sw(F(21), ws + WSO_3C, 512, 768, 512);

  pcd_encoder_kernel<<<nobj, NTHREADS, SMEM_BYTES, stream>>>(
      pc,
      ws + WSO_1A, ws + WSO_1B, ws + WSO_1C,
      F(4),  F(5),  F(6),   // sa1_s
      F(7),  F(8),  F(9),   // sa1_b
      ws + WSO_2A, ws + WSO_2B, ws + WSO_2C,
      F(13), F(14), F(15),  // sa2_s
      F(16), F(17), F(18),  // sa2_b
      ws + WSO_3A, ws + WSO_3B, ws + WSO_3C,
      F(22), F(23), F(24),  // sa3_s
      F(25), F(26), F(27),  // sa3_b
      F(28), F(29),         // fc_w, fc_b
      (float*)d_out);
}
